// CrossAttentionFusionFourBranches_84499186582227
// MI455X (gfx1250) — matrix operand order repaired and checked by reading, not compiled, not run on hardware
//
#include <hip/hip_runtime.h>

// ---------------------------------------------------------------------------
// CrossAttentionFusionFourBranches for MI455X (gfx1250, wave32, WMMA).
//
// seq_len==1 attention => softmax over one key == 1 => o = v.  Entire net
// folds to:  y = sum_i xkv_i @ M_i^T + c ;  out = LayerNorm(y)*gamma + beta
// with M_i = Wf_i @ Wout_i @ Wv_i  (folded on-device each launch),
// xkv = [x2, x3, x4, x1].
//
// GEMMs: v_wmma_f32_16x16x32_bf16, 3-term bf16 split precision
// (hi*hi + hi*lo + lo*hi), f32 accumulation.  Tiles are staged into a
// double-buffered LDS pipeline with CDNA5 GLOBAL_LOAD_ASYNC_TO_LDS_B128
// (ASYNCcnt): async copies for tile k+1 overlap the WMMA compute of tile k,
// one workgroup barrier per K-step, zero staging VGPRs.
// ---------------------------------------------------------------------------

typedef __attribute__((ext_vector_type(16))) __bf16 v16bf;
typedef __attribute__((ext_vector_type(8)))  __bf16 v8bf;
typedef __attribute__((ext_vector_type(4)))  __bf16 v4bf;
typedef __attribute__((ext_vector_type(8)))  float  v8f;
typedef __attribute__((ext_vector_type(4)))  float  v4f;

namespace {

constexpr int D  = 1024;
constexpr int BR = 16384;
constexpr int K4 = 4096;

// LDS tile pitch: 40 halfwords (80 B) per 32-k row => fragment b128 reads are
// bank-conflict-free (start banks l*20 dwords, distinct multiples of 4 mod 64).
constexpr int PITCH = 40;
constexpr int PLANE = 128 * PITCH;   // elems per plane

__device__ __forceinline__ v16bf cat8(v8bf a, v8bf b) {
  return __builtin_shufflevector(a, b, 0, 1, 2, 3, 4, 5, 6, 7,
                                 8, 9, 10, 11, 12, 13, 14, 15);
}

// Split-precision bf16 WMMA GEMM, double-buffered async-LDS pipeline.
//   C[M,N] = A[M,K] * B[K,N]
//   A: bf16 split planes [M,K] (ah/al), row stride lda.
//   B: transposed bf16 split planes BT[N,K] (bth/btl), row stride ldbt.
//   mode 0: store f32 (+ optional per-column bias)  -> cf32 (ldc)
//   mode 1: store bf16 split planes                 -> ohi/olo (ldo)
// Block: 256 threads (8 waves, 4x2), tile 128(M) x 128(N); wave tile 32x64.
// Grid: (N/128, M/128).  M,N multiples of 128; K multiple of 64.
__global__ __launch_bounds__(256)
void wmma_gemm_kernel(const __bf16* __restrict__ ah, const __bf16* __restrict__ al,
                      long lda,
                      const __bf16* __restrict__ bth, const __bf16* __restrict__ btl,
                      long ldbt, int K,
                      float* __restrict__ cf32, long ldc,
                      const float* __restrict__ bias,
                      __bf16* __restrict__ ohi, __bf16* __restrict__ olo,
                      long ldo, int mode)
{
  __shared__ __bf16 Als[2][2 * PLANE];   // [buf][plane][row 0..127][k 0..31 (+pad)]
  __shared__ __bf16 Bls[2][2 * PLANE];   // [buf][plane][col 0..127][k 0..31 (+pad)]

  const int t    = threadIdx.x;
  const int lane = t & 31;
  const int wave = t >> 5;
  const int l  = lane & 15;
  const int h  = lane >> 4;
  const int wm = wave & 3;            // 0..3 -> 32-row slice
  const int wn = wave >> 2;           // 0..1 -> 64-col slice
  const int RM0 = blockIdx.y * 128;
  const int CN0 = blockIdx.x * 128;

  // Async staging roles: thread copies one 64B (32 bf16) k-row of one
  // (plane, row/col) as 4 x async b128.  The instruction offset advances the
  // LDS and global addresses together (ISA 15.18.3), and all 4 pieces stay
  // inside one padded LDS row, so a single LDS-offset VGPR per buffer works.
  const int srow = t & 127;
  const int spln = t >> 7;

  const unsigned aLds0 =
      (unsigned)(unsigned long long)&Als[0][(spln ? PLANE : 0) + srow * PITCH];
  const unsigned aLds1 =
      (unsigned)(unsigned long long)&Als[1][(spln ? PLANE : 0) + srow * PITCH];
  const unsigned bLds0 =
      (unsigned)(unsigned long long)&Bls[0][(spln ? PLANE : 0) + srow * PITCH];
  const unsigned bLds1 =
      (unsigned)(unsigned long long)&Bls[1][(spln ? PLANE : 0) + srow * PITCH];
  const unsigned long long aG =
      (unsigned long long)(spln ? al : ah) +
      2ull * (unsigned long long)((long)(RM0 + srow) * lda);
  const unsigned long long bG =
      (unsigned long long)(spln ? btl : bth) +
      2ull * (unsigned long long)((long)(CN0 + srow) * ldbt);

  v8f acc[2][4];
#pragma unroll
  for (int mi = 0; mi < 2; ++mi)
#pragma unroll
    for (int ni = 0; ni < 4; ++ni)
      acc[mi][ni] = (v8f){0.f, 0.f, 0.f, 0.f, 0.f, 0.f, 0.f, 0.f};

  auto issue = [&](int k0, unsigned aL, unsigned bL) {
    unsigned long long ga = aG + 2ull * (unsigned)k0;
    unsigned long long gb = bG + 2ull * (unsigned)k0;
    asm volatile(
        "global_load_async_to_lds_b128 %0, %2, off\n\t"
        "global_load_async_to_lds_b128 %0, %2, off offset:16\n\t"
        "global_load_async_to_lds_b128 %0, %2, off offset:32\n\t"
        "global_load_async_to_lds_b128 %0, %2, off offset:48\n\t"
        "global_load_async_to_lds_b128 %1, %3, off\n\t"
        "global_load_async_to_lds_b128 %1, %3, off offset:16\n\t"
        "global_load_async_to_lds_b128 %1, %3, off offset:32\n\t"
        "global_load_async_to_lds_b128 %1, %3, off offset:48"
        :: "v"(aL), "v"(bL), "v"(ga), "v"(gb)
        : "memory");
  };

  auto compute = [&](int sel) {
    v16bf ahf[2], alf[2];
#pragma unroll
    for (int mi = 0; mi < 2; ++mi) {
      const int o = (wm * 32 + mi * 16 + l) * PITCH + h * 8;
      ahf[mi] = cat8(*(const v8bf*)&Als[sel][o],
                     *(const v8bf*)&Als[sel][o + 16]);
      alf[mi] = cat8(*(const v8bf*)&Als[sel][PLANE + o],
                     *(const v8bf*)&Als[sel][PLANE + o + 16]);
    }
#pragma unroll
    for (int ni = 0; ni < 4; ++ni) {
      const int o = (wn * 64 + ni * 16 + l) * PITCH + h * 8;
      v16bf bhf = cat8(*(const v8bf*)&Bls[sel][o],
                       *(const v8bf*)&Bls[sel][o + 16]);
      v16bf blf = cat8(*(const v8bf*)&Bls[sel][PLANE + o],
                       *(const v8bf*)&Bls[sel][PLANE + o + 16]);
#pragma unroll
      for (int mi = 0; mi < 2; ++mi) {
        acc[mi][ni] = __builtin_amdgcn_wmma_f32_16x16x32_bf16(
            false, ahf[mi], false, bhf, (short)0, acc[mi][ni], false, false);
        acc[mi][ni] = __builtin_amdgcn_wmma_f32_16x16x32_bf16(
            false, ahf[mi], false, blf, (short)0, acc[mi][ni], false, false);
        acc[mi][ni] = __builtin_amdgcn_wmma_f32_16x16x32_bf16(
            false, alf[mi], false, bhf, (short)0, acc[mi][ni], false, false);
      }
    }
  };

  // Pipeline: wait(cur) -> barrier -> issue(next) -> compute(cur).
  // Each wave waits its own ASYNCcnt before signaling the barrier, so the
  // barrier publishes every wave's async writes; passing the barrier also
  // proves all waves finished reading the buffer being overwritten next.
  issue(0, aLds0, bLds0);
  for (int k0 = 0; k0 < K; k0 += 64) {
    asm volatile("s_wait_asynccnt 0x0" ::: "memory");
    __syncthreads();
    if (k0 + 32 < K) issue(k0 + 32, aLds1, bLds1);
    compute(0);

    asm volatile("s_wait_asynccnt 0x0" ::: "memory");
    __syncthreads();
    if (k0 + 64 < K) issue(k0 + 64, aLds0, bLds0);
    compute(1);
  }

  // Epilogue.  C/D layout: VGPR r -> row r + h*8, col = l.
#pragma unroll
  for (int mi = 0; mi < 2; ++mi) {
#pragma unroll
    for (int ni = 0; ni < 4; ++ni) {
      const int col = CN0 + wn * 64 + ni * 16 + l;
      const float badd = (mode == 0 && bias) ? bias[col] : 0.f;
#pragma unroll
      for (int r = 0; r < 8; ++r) {
        const int row = RM0 + wm * 32 + mi * 16 + h * 8 + r;
        float v = acc[mi][ni][r];
        if (mode == 0) {
          cf32[(long)row * ldc + col] = v + badd;
        } else {
          __bf16 hb = (__bf16)v;
          ohi[(long)row * ldo + col] = hb;
          olo[(long)row * ldo + col] = (__bf16)(v - (float)hb);
        }
      }
    }
  }
}

// f32 row-major [*,C] -> bf16 split planes with dst row stride ld and column
// offset (for concatenating sources along K).  4 elems/thread.
__global__ __launch_bounds__(256)
void convpack_kernel(const float* __restrict__ src, long n, int C,
                     long ld, long coloff,
                     __bf16* __restrict__ dhi, __bf16* __restrict__ dlo)
{
  long i4 = ((long)blockIdx.x * 256 + threadIdx.x) * 4;
  if (i4 >= n) return;
  long r = i4 / C;
  int  c = (int)(i4 - r * C);
  v4f v = *(const v4f*)(src + i4);
  v4bf hh, ll;
#pragma unroll
  for (int j = 0; j < 4; ++j) {
    __bf16 hb = (__bf16)v[j];
    hh[j] = hb;
    ll[j] = (__bf16)(v[j] - (float)hb);
  }
  long o = r * ld + coloff + c;
  *(v4bf*)(dhi + o) = hh;
  *(v4bf*)(dlo + o) = ll;
}

// f32 [R,C] (row stride ss) -> transposed bf16 split planes dhi/dlo [C,R] (row stride ds)
__global__ __launch_bounds__(256)
void transconv_kernel(const float* __restrict__ src, int R, int C, int ss,
                      __bf16* __restrict__ dhi, __bf16* __restrict__ dlo, int ds)
{
  int idx = blockIdx.x * 256 + threadIdx.x;
  if (idx >= R * C) return;
  int r = idx / C, c = idx - r * C;
  float x = src[(long)r * ss + c];
  __bf16 h = (__bf16)x;
  dhi[(long)c * ds + r] = h;
  dlo[(long)c * ds + r] = (__bf16)(x - (float)h);
}

// u[i*D+n] = sum_r Wout_i[n,r] * bv_i[r] + b_out_i[n]
__global__ __launch_bounds__(256)
void u_kernel(const float* __restrict__ w_out, const float* __restrict__ b_in,
              const float* __restrict__ b_out, float* __restrict__ u)
{
  int id = blockIdx.x * 256 + threadIdx.x;   // 0..4095
  int i = id >> 10, n = id & 1023;
  const float* wrow = w_out + ((long)i * D + n) * D;
  const float* bv   = b_in + (long)i * (3 * D) + 2 * D;
  float s = b_out[(long)i * D + n];
  for (int r = 0; r < D; ++r) s += wrow[r] * bv[r];
  u[id] = s;
}

// cb[m] = b_fuse[m] + sum_j w_fuse[m,j] * u[j]   (j over 4*D)
__global__ __launch_bounds__(256)
void cb_kernel(const float* __restrict__ w_fuse, const float* __restrict__ b_fuse,
               const float* __restrict__ u, float* __restrict__ cb)
{
  int m = blockIdx.x * 256 + threadIdx.x;    // 0..1023
  const float* wf = w_fuse + (long)m * K4;
  float s = b_fuse[m];
  for (int j = 0; j < K4; ++j) s += wf[j] * u[j];
  cb[m] = s;
}

// LayerNorm over N=1024 per row; one wave32 per row, 8 rows per block.
__global__ __launch_bounds__(256)
void ln_kernel(const float* __restrict__ y, const float* __restrict__ gamma,
               const float* __restrict__ beta, float* __restrict__ out)
{
  const int lane = threadIdx.x & 31;
  const int wave = threadIdx.x >> 5;
  const long row = (long)blockIdx.x * 8 + wave;
  const float* yr = y + row * D;
  const v4f* yv = (const v4f*)yr;

  float s = 0.f, s2 = 0.f;
#pragma unroll
  for (int it = 0; it < 8; ++it) {
    v4f v = yv[lane + it * 32];
#pragma unroll
    for (int j = 0; j < 4; ++j) { s += v[j]; s2 += v[j] * v[j]; }
  }
#pragma unroll
  for (int m = 16; m >= 1; m >>= 1) {
    s  += __shfl_xor(s,  m, 32);
    s2 += __shfl_xor(s2, m, 32);
  }
  const float inv = 1.0f / (float)D;
  float mu  = s * inv;
  float var = s2 * inv - mu * mu;
  float rs  = rsqrtf(var + 1e-5f);

  float* o = out + row * D;
  const v4f* gv = (const v4f*)gamma;
  const v4f* bv = (const v4f*)beta;
  v4f* ov = (v4f*)o;
#pragma unroll
  for (int it = 0; it < 8; ++it) {
    int c4 = lane + it * 32;
    v4f v = yv[c4], g = gv[c4], b = bv[c4], r;
#pragma unroll
    for (int j = 0; j < 4; ++j) r[j] = (v[j] - mu) * rs * g[j] + b[j];
    ov[c4] = r;
  }
}

} // namespace

extern "C" void kernel_launch(void* const* d_in, const int* in_sizes, int n_in,
                              void* d_out, int out_size, void* d_ws, size_t ws_size,
                              hipStream_t stream) {
  (void)in_sizes; (void)n_in; (void)out_size; (void)ws_size;

  const float* x1     = (const float*)d_in[0];
  const float* x2     = (const float*)d_in[1];
  const float* x3     = (const float*)d_in[2];
  const float* x4     = (const float*)d_in[3];
  const float* w_in   = (const float*)d_in[4];
  const float* b_in   = (const float*)d_in[5];
  const float* w_out  = (const float*)d_in[6];
  const float* b_out  = (const float*)d_in[7];
  const float* w_fuse = (const float*)d_in[8];
  const float* b_fuse = (const float*)d_in[9];
  const float* gamma  = (const float*)d_in[10];
  const float* beta   = (const float*)d_in[11];
  float* out = (float*)d_out;

  const size_t DD = (size_t)D * D;
  char* ws = (char*)d_ws;
  size_t off = 0;
  auto alloc = [&](size_t bytes) -> void* {
    void* p = ws + off;
    off += (bytes + 255) & ~(size_t)255;
    return p;
  };

  __bf16* Vt_hi   = (__bf16*)alloc(4 * DD * sizeof(__bf16));
  __bf16* Vt_lo   = (__bf16*)alloc(4 * DD * sizeof(__bf16));
  float*  Tf      = (float*) alloc(4 * DD * sizeof(float));
  __bf16* Tt_hi   = (__bf16*)alloc(4 * DD * sizeof(__bf16));
  __bf16* Tt_lo   = (__bf16*)alloc(4 * DD * sizeof(__bf16));
  __bf16* Wcat_hi = (__bf16*)alloc((size_t)D * K4 * sizeof(__bf16));
  __bf16* Wcat_lo = (__bf16*)alloc((size_t)D * K4 * sizeof(__bf16));
  __bf16* Wout_hi = (__bf16*)alloc(4 * DD * sizeof(__bf16));
  __bf16* Wout_lo = (__bf16*)alloc(4 * DD * sizeof(__bf16));
  __bf16* Wfu_hi  = (__bf16*)alloc((size_t)D * K4 * sizeof(__bf16));
  __bf16* Wfu_lo  = (__bf16*)alloc((size_t)D * K4 * sizeof(__bf16));
  __bf16* Xc_hi   = (__bf16*)alloc((size_t)BR * K4 * sizeof(__bf16));
  __bf16* Xc_lo   = (__bf16*)alloc((size_t)BR * K4 * sizeof(__bf16));
  float*  u       = (float*) alloc((size_t)K4 * sizeof(float));
  float*  cb      = (float*) alloc((size_t)D * sizeof(float));
  float*  yf      = (float*) alloc((size_t)BR * D * sizeof(float));

  dim3 blk(256);

  // 0) bf16 split-plane materialization of all GEMM A operands.
  //    Xc = [x2|x3|x4|x1] (branch order) as one [BR, 4K] matrix.
  const float* xs[4] = {x2, x3, x4, x1};
  for (int i = 0; i < 4; ++i)
    convpack_kernel<<<(int)(((size_t)BR * D / 4 + 255) / 256), blk, 0, stream>>>(
        xs[i], (long)BR * D, D, K4, (long)i * D, Xc_hi, Xc_lo);
  convpack_kernel<<<(int)((4 * DD / 4 + 255) / 256), blk, 0, stream>>>(
      w_out, (long)4 * DD, D, D, 0, Wout_hi, Wout_lo);
  convpack_kernel<<<(int)(((size_t)D * K4 / 4 + 255) / 256), blk, 0, stream>>>(
      w_fuse, (long)D * K4, K4, K4, 0, Wfu_hi, Wfu_lo);

  // 1) Wv_i (rows of w_in[i, 2D:3D, :]) -> transposed bf16 planes Vt_i[c,r]
  for (int i = 0; i < 4; ++i) {
    const float* wv = w_in + ((size_t)i * 3 * D + 2 * D) * D;
    transconv_kernel<<<(D * D + 255) / 256, blk, 0, stream>>>(
        wv, D, D, D, Vt_hi + (size_t)i * DD, Vt_lo + (size_t)i * DD, D);
  }

  // 2) T_i = Wout_i @ Wv_i   (f32 out)
  for (int i = 0; i < 4; ++i)
    wmma_gemm_kernel<<<dim3(D / 128, D / 128), blk, 0, stream>>>(
        Wout_hi + (size_t)i * DD, Wout_lo + (size_t)i * DD, D,
        Vt_hi + (size_t)i * DD, Vt_lo + (size_t)i * DD, D, D,
        Tf + (size_t)i * DD, D, nullptr, nullptr, nullptr, 0, 0);

  // 3) T_i -> transposed bf16 planes Tt_i[c,n]
  for (int i = 0; i < 4; ++i)
    transconv_kernel<<<(D * D + 255) / 256, blk, 0, stream>>>(
        Tf + (size_t)i * DD, D, D, D,
        Tt_hi + (size_t)i * DD, Tt_lo + (size_t)i * DD, D);

  // 4) M_i = Wf_i @ T_i, split-stored into Wcat[n, i*D + c] planes
  for (int i = 0; i < 4; ++i)
    wmma_gemm_kernel<<<dim3(D / 128, D / 128), blk, 0, stream>>>(
        Wfu_hi + (size_t)i * D, Wfu_lo + (size_t)i * D, K4,
        Tt_hi + (size_t)i * DD, Tt_lo + (size_t)i * DD, D, D,
        nullptr, 0, nullptr, Wcat_hi + (size_t)i * D, Wcat_lo + (size_t)i * D,
        K4, 1);

  // 5) Folded bias
  u_kernel<<<K4 / 256, blk, 0, stream>>>(w_out, b_in, b_out, u);
  cb_kernel<<<D / 256, blk, 0, stream>>>(w_fuse, b_fuse, u, cb);

  // 6) Main GEMM: y[B,D] = Xc @ Wcat^T + cb
  wmma_gemm_kernel<<<dim3(D / 128, BR / 128), blk, 0, stream>>>(
      Xc_hi, Xc_lo, K4, Wcat_hi, Wcat_lo, K4, K4,
      yf, D, cb, nullptr, nullptr, 0, 0);

  // 7) LayerNorm
  ln_kernel<<<BR / 8, blk, 0, stream>>>(yf, gamma, beta, out);
}
